// OEG_CFN_54949811585130
// MI455X (gfx1250) — matrix operand order repaired and checked by reading, not compiled, not run on hardware
//
#include <hip/hip_runtime.h>
#include <hip/hip_bf16.h>
#include <math.h>

// ---------------------------------------------------------------------------
// Local correlation (R=6 -> 169 displacements, 128-ch dot) + softmax warp.
// B=4, C=64 (x2 concat -> 128), H=W=256.
//
// Block = 256 threads (8 waves). Tile = 8 query rows x 16 query cols.
// LDS: query tile (bf16), zero-padded key window 20x32x128 (bf16),
//      image window 20x32 (f32), per-wave 16x32 logit scratch (f32).
// Each wave: A-fragments preloaded in registers; per dy: 2 key tiles x 4
// chained v_wmma_f32_16x16x32_bf16 (K=128), spill logits to LDS, then a
// lane-split (7/6 diagonals per lane-half) online softmax with one rescale
// exp per dy; halves merged at the end via ds_swizzle_b32 SWAPX16.
// ---------------------------------------------------------------------------

typedef __attribute__((ext_vector_type(16))) __bf16 v16bf;
typedef __attribute__((ext_vector_type(8)))  __bf16 v8bf;
typedef __attribute__((ext_vector_type(8)))  float  v8f;

namespace {
constexpr int Hh = 256, Ww = 256, Cc = 64;   // per-input channel count
constexpr int Rr = 6;                         // displacement radius
constexpr int Dd = 2 * Rr + 1;                // 13
constexpr int ROWS = 8;                       // query rows per block
constexpr int QW = 16;                        // query cols per block
constexpr int KR = ROWS + 2 * Rr;             // 20 key rows staged
constexpr int KW = 32;                        // key cols staged (2 tiles)
constexpr int C2 = 128;                       // concat channels
constexpr int CP = 136;                       // padded channel stride (272 B rows)
constexpr int CORRW = 33;                     // padded logit row stride

constexpr size_t SQ_ELEMS   = (size_t)ROWS * QW * CP;     // bf16
constexpr size_t SK_ELEMS   = (size_t)KR * KW * CP;       // bf16
constexpr size_t SIMG_ELEMS = (size_t)KR * KW;            // f32
constexpr size_t SCORR_ELEMS = (size_t)ROWS * 16 * CORRW; // f32
constexpr size_t SMEM_BYTES =
    SQ_ELEMS * 2 + SK_ELEMS * 2 + SIMG_ELEMS * 4 + SCORR_ELEMS * 4; // 228352
}

__device__ __forceinline__ __bf16 f2bf(float f) {
  return (__bf16)f;                 // hardware f32->bf16 convert (RTNE)
}

// Exchange a float between lane i and lane i^16 (wave32): ds_swizzle SWAPX16.
__device__ __forceinline__ float swap16(float x) {
  int xi = __builtin_bit_cast(int, x);
  int yi = __builtin_amdgcn_ds_swizzle(xi, 0x401F); // xor=0x10, and=0x1f
  return __builtin_bit_cast(float, yi);
}

__global__ __launch_bounds__(256)
void corr_softmax_warp_kernel(const float* __restrict__ feats_r,
                              const float* __restrict__ feats_t,
                              const float* __restrict__ img_r,
                              const float* __restrict__ plf_1,
                              const float* __restrict__ plf_2,
                              float* __restrict__ out) {
  extern __shared__ __align__(16) char smem_raw[];
  __bf16* sQ   = (__bf16*)smem_raw;                 // [ROWS*QW][CP]
  __bf16* sK   = sQ + SQ_ELEMS;                     // [KR*KW][CP]
  float*  sImg = (float*)(sK + SK_ELEMS);           // [KR][KW]
  float*  sCorr = sImg + SIMG_ELEMS;                // [ROWS][16][CORRW]

  const int tid = threadIdx.x;
  const int b  = blockIdx.z;
  const int h0 = blockIdx.y * ROWS;
  const int w0 = blockIdx.x * QW;

  // ---- cooperative staging -------------------------------------------------
  // Query tile: x1 = concat(feats_t, plf_1). Always in-bounds.
  for (int e = tid; e < ROWS * QW * C2; e += 256) {
    const int qrow = e >> 11;            // /(128*16)
    const int c    = (e >> 4) & 127;
    const int px   = e & 15;
    const float* src = (c < Cc) ? feats_t : plf_1;
    const float v = src[(((size_t)b * Cc + (c & 63)) * Hh + (h0 + qrow)) * Ww +
                        (w0 + px)];
    sQ[((size_t)(qrow * QW + px)) * CP + c] = f2bf(v);
  }
  // Key window: x2 = concat(feats_r, plf_2), zero-padded (matches reference
  // zero-pad: OOB logits become exactly 0 and still enter the softmax).
  for (int e = tid; e < KR * KW * C2; e += 256) {
    const int kr = e >> 12;              // /(128*32)
    const int c  = (e >> 5) & 127;
    const int kx = e & 31;
    const int y = h0 + kr - Rr;
    const int x = w0 + kx - Rr;
    float v = 0.0f;
    if (y >= 0 && y < Hh && x >= 0 && x < Ww) {
      const float* src = (c < Cc) ? feats_r : plf_2;
      v = src[(((size_t)b * Cc + (c & 63)) * Hh + y) * Ww + x];
    }
    sK[((size_t)(kr * KW + kx)) * CP + c] = f2bf(v);
  }
  // Image window (zero-padded).
  for (int e = tid; e < KR * KW; e += 256) {
    const int kr = e >> 5, kx = e & 31;
    const int y = h0 + kr - Rr, x = w0 + kx - Rr;
    sImg[e] = (y >= 0 && y < Hh && x >= 0 && x < Ww)
                  ? img_r[((size_t)b * Hh + y) * Ww + x] : 0.0f;
  }
  __syncthreads();

  // ---- per-wave compute: one query row per wave ----------------------------
  const int wave = tid >> 5;
  const int lane = tid & 31;
  const int qrow = wave;
  const int Mi = lane & 15;        // M (query col) for A / N (key col) for B
  const int hi = lane >> 4;        // lane-half selector per 16-bit layouts

  const __bf16* qbase = sQ + ((size_t)(qrow * QW + Mi)) * CP;
  float* cw = sCorr + (size_t)qrow * 16 * CORRW;
  const float* crow = cw + Mi * CORRW + Mi;   // this lane's diagonal base

  // Preload A fragments: invariant across all 13 dy iterations.
  // (16-bit A 16x32 layout: lane<16 -> K {0..7,16..23}, lane>=16 ->
  //  K {8..15,24..31}, all +32*kc.)
  v16bf Af[4];
  #pragma unroll
  for (int kc = 0; kc < 4; ++kc) {
    const int c0 = kc * 32 + hi * 8;
    const v8bf a0 = *(const v8bf*)(qbase + c0);
    const v8bf a1 = *(const v8bf*)(qbase + c0 + 16);
    #pragma unroll
    for (int e2 = 0; e2 < 8; ++e2) { Af[kc][e2] = a0[e2]; Af[kc][e2 + 8] = a1[e2]; }
  }

  // Online softmax state; lane-half hi handles diagonals dx = 7*hi .. 7*hi+6
  // (hi=1 has only 6 valid; the 7th is masked to -inf / 0).
  const int dxBase = hi * 7;
  float m = -INFINITY, ssum = 0.0f, accw = 0.0f;

  for (int dy = 0; dy < Dd; ++dy) {
    const int kr = qrow + dy;        // key row index in staged window
    #pragma unroll
    for (int t = 0; t < 2; ++t) {    // two 16-key tiles cover 13 diagonals
      v8f acc = {};
      const __bf16* kbase =
          sK + ((size_t)(kr * KW + t * 16 + Mi)) * CP + hi * 8;
      #pragma unroll
      for (int kc = 0; kc < 4; ++kc) {                 // K = 128 = 4 x 32
        // B fragment (16-bit B 32x16): lane holds column N=lane&15,
        // K = {hi*8 + 0..7, 16 + hi*8 + 0..7} (+32*kc), matching A.
        const v8bf b0 = *(const v8bf*)(kbase + kc * 32);
        const v8bf b1 = *(const v8bf*)(kbase + kc * 32 + 16);
        v16bf Bm;
        #pragma unroll
        for (int e2 = 0; e2 < 8; ++e2) { Bm[e2] = b0[e2]; Bm[e2 + 8] = b1[e2]; }
        acc = __builtin_amdgcn_wmma_f32_16x16x32_bf16(
            false, Af[kc], false, Bm, (short)0, acc, false, false);
      }
      // C/D layout: lanes 0-15 VGPRr -> M=r; lanes 16-31 VGPRr -> M=8+r.
      #pragma unroll
      for (int r = 0; r < 8; ++r)
        cw[(hi * 8 + r) * CORRW + t * 16 + Mi] = acc[r];
    }
    asm volatile("s_wait_dscnt 0" ::: "memory");  // logits visible wave-wide

    // Gather this lane-half's logits / image taps for this dy.
    const float* irow = sImg + kr * KW + Mi;
    float cv[7], iw[7];
    float mloc = -INFINITY;
    #pragma unroll
    for (int k = 0; k < 7; ++k) {
      const int dx = dxBase + k;
      const bool valid = dx < Dd;
      const float c_ = valid ? crow[dx] : -INFINITY;
      cv[k] = c_;
      iw[k] = valid ? irow[dx] : 0.0f;
      mloc = fmaxf(mloc, c_);
    }
    // One rescale exp per dy, one exp per logit.
    const float mn = fmaxf(m, mloc);
    const float sc = __expf(m - mn);     // 0 on first iteration (m=-inf)
    ssum *= sc;
    accw *= sc;
    #pragma unroll
    for (int k = 0; k < 7; ++k) {
      const float ev = __expf(cv[k] - mn);   // 0 for masked slot
      ssum += ev;
      accw += ev * iw[k];
    }
    m = mn;
  }

  // Merge the two lane-half partial softmaxes (lane i <-> lane i+16).
  const float m2 = swap16(m);
  const float s2 = swap16(ssum);
  const float a2 = swap16(accw);
  const float mt = fmaxf(m, m2);
  const float e1 = __expf(m - mt), e2 = __expf(m2 - mt);
  const float S  = ssum * e1 + s2 * e2;
  const float Aw = accw * e1 + a2 * e2;

  if (hi == 0)
    out[((size_t)b * Hh + (h0 + qrow)) * Ww + (w0 + Mi)] = Aw / S;
}

extern "C" void kernel_launch(void* const* d_in, const int* in_sizes, int n_in,
                              void* d_out, int out_size, void* d_ws, size_t ws_size,
                              hipStream_t stream) {
  const float* feats_r = (const float*)d_in[0];
  const float* feats_t = (const float*)d_in[1];
  const float* img_r   = (const float*)d_in[2];
  const float* plf_1   = (const float*)d_in[3];
  const float* plf_2   = (const float*)d_in[4];
  float* out = (float*)d_out;

  (void)in_sizes; (void)n_in; (void)out_size; (void)d_ws; (void)ws_size;

  // Allow >64KB dynamic LDS (320KB/WGP on CDNA5). Idempotent, capture-safe.
  hipFuncSetAttribute((const void*)corr_softmax_warp_kernel,
                      hipFuncAttributeMaxDynamicSharedMemorySize,
                      (int)SMEM_BYTES);

  dim3 grid(Ww / QW, Hh / ROWS, 4);   // 16 x 32 x 4 = 2048 blocks
  dim3 block(256);                    // 8 waves: one query row each
  corr_softmax_warp_kernel<<<grid, block, SMEM_BYTES, stream>>>(
      feats_r, feats_t, img_r, plf_1, plf_2, out);
}
